// HGTModel_28432683500202
// MI455X (gfx1250) — matrix operand order repaired and checked by reading, not compiled, not run on hardware
//
#include <hip/hip_runtime.h>
#include <hip/hip_bf16.h>
#include <math.h>

#define HID   128
#define HEADS 8
#define DH    16
#define LNUM  2
#define OUTD  32
#define SQRTD 4.0f
#define LN_EPS 1e-5f

typedef __attribute__((ext_vector_type(16))) __bf16 bf16x16;
typedef __attribute__((ext_vector_type(8)))  float  f32x8;

__device__ __forceinline__ float sigmoidf_(float x) { return 1.0f / (1.0f + __expf(-x)); }
__device__ __forceinline__ float gelu_exact(float x) { return 0.5f * x * (1.0f + erff(x * 0.70710678118654752f)); }

__device__ __forceinline__ void atomicMaxF(float* addr, float val) {
    int old = __float_as_int(*addr);
    while (__int_as_float(old) < val) {
        int assumed = old;
        old = atomicCAS((int*)addr, assumed, __float_as_int(val));
        if (old == assumed) break;
    }
}

// ---------------------------------------------------------------------------
// C[M,N] = act(A[M,128] @ B[128,N] + bias[N])   (act: 0=none, 1=relu)
// One wave computes one 16x16 output tile using v_wmma_f32_16x16x32_bf16.
// Fragment layouts per CDNA5 ISA 7.12.2 (wave32):
//   A 16x32 bf16: lane -> M=lane&15; vgpr v, halves -> K = 16*(v>=4) + 8*(lane>=16) + 2*(v&3) + half
//   B 32x16 bf16: lane -> N=lane&15; element j -> K = 16*(j>=8) + 8*(lane>=16) + (j&7)
//   C/D f32:      vgpr r, lane -> M = r + 8*(lane>=16), N = lane&15
// ---------------------------------------------------------------------------
__global__ void gemm_bias_act(const float* __restrict__ A, const float* __restrict__ B,
                              const float* __restrict__ bias, float* __restrict__ C,
                              int M, int N, int act)
{
    const int lane = threadIdx.x & 31;
    const int wave = (blockIdx.x * blockDim.x + threadIdx.x) >> 5;
    const int tilesN = N >> 4;
    const int tiles  = (M >> 4) * tilesN;
    if (wave >= tiles) return;                   // wave-uniform: EXEC stays all-1 for WMMA
    const int tm = wave / tilesN;
    const int tn = wave - tm * tilesN;
    const int m0 = tm << 4;
    const int n0 = tn << 4;

    const int mrow = lane & 15;
    const int koff = (lane < 16) ? 0 : 8;
    const float* Arow = A + (size_t)(m0 + mrow) * HID;

    f32x8 acc = {};
    #pragma unroll
    for (int kk = 0; kk < HID; kk += 32) {
        bf16x16 af, bfv;
        #pragma unroll
        for (int vg = 0; vg < 8; ++vg) {
            const int k = kk + ((vg & 4) << 2) + koff + ((vg & 3) << 1);
            const float2 f2 = *(const float2*)(Arow + k);
            af[2 * vg]     = (__bf16)f2.x;
            af[2 * vg + 1] = (__bf16)f2.y;
        }
        const float* Brow = B + (size_t)kk * N + n0 + (lane & 15);
        #pragma unroll
        for (int j = 0; j < 16; ++j) bfv[j] = (__bf16)Brow[(size_t)(((j & 8) << 1) + koff + (j & 7)) * N];
        acc = __builtin_amdgcn_wmma_f32_16x16x32_bf16(false, af, false, bfv,
                                                      (short)0, acc, false, false);
    }

    const int n = lane & 15;
    const float bv = bias ? bias[n0 + n] : 0.0f;
    const int mbase = m0 + ((lane >> 4) << 3);
    #pragma unroll
    for (int r = 0; r < 8; ++r) {
        float v = acc[r] + bv;
        if (act == 1) v = fmaxf(v, 0.0f);
        C[(size_t)(mbase + r) * N + n0 + n] = v;
    }
}

// ---------------------------------------------------------------------------
// Fold per-head relation matrix (and optional p_rel/sqrt(D) scale) into weights:
//   Wdst[i, h*16+e] = sum_d Wsrc[i, h*16+d] * rel[h,d,e] * scale(h)
//   row 128 computes the fused bias.
// ---------------------------------------------------------------------------
__global__ void fuse_rel(const float* __restrict__ Wsrc, const float* __restrict__ bsrc,
                         const float* __restrict__ rel, const float* __restrict__ pr,
                         float* __restrict__ Wdst, float* __restrict__ bdst)
{
    int idx = blockIdx.x * blockDim.x + threadIdx.x;
    if (idx >= 129 * HID) return;
    int r = idx >> 7;        // 0..128 (128 == bias row)
    int c = idx & 127;
    int h = c >> 4, e = c & 15;
    float scale = pr ? pr[h] * (1.0f / SQRTD) : 1.0f;
    const float* relh = rel + h * DH * DH;
    float s = 0.0f;
    if (r < HID) {
        const float* wr = Wsrc + (size_t)r * HID + h * DH;
        #pragma unroll
        for (int d = 0; d < DH; ++d) s += wr[d] * relh[d * DH + e];
        Wdst[(size_t)r * HID + c] = s * scale;
    } else {
        const float* br = bsrc + h * DH;
        #pragma unroll
        for (int d = 0; d < DH; ++d) s += br[d] * relh[d * DH + e];
        bdst[c] = s * scale;
    }
}

__global__ void fill_kernel(float* __restrict__ p, float v, int n)
{
    int i = blockIdx.x * blockDim.x + threadIdx.x;
    if (i < n) p[i] = v;
}

// logit[e,h] = q[dst[e],h,:] . kt[src[e],h,:]   (p/sqrtD pre-folded into kt)
__global__ void edge_logits(const int* __restrict__ src, const int* __restrict__ dst,
                            const float* __restrict__ q, const float* __restrict__ kt,
                            float* __restrict__ logits, float* __restrict__ mx, int E_)
{
    int idx = blockIdx.x * blockDim.x + threadIdx.x;
    if (idx >= E_ * HEADS) return;
    int e = idx >> 3, h = idx & 7;
    int s = src[e], d = dst[e];
    const float4* qp = (const float4*)(q  + (size_t)d * HID + h * DH);
    const float4* kp = (const float4*)(kt + (size_t)s * HID + h * DH);
    float acc = 0.0f;
    #pragma unroll
    for (int i = 0; i < 4; ++i) {
        float4 a = qp[i], b = kp[i];
        acc += a.x * b.x + a.y * b.y + a.z * b.z + a.w * b.w;
    }
    logits[idx] = acc;
    atomicMaxF(mx + (size_t)d * HEADS + h, acc);
}

__global__ void edge_expsum(const int* __restrict__ dst, float* __restrict__ logits,
                            const float* __restrict__ mx, float* __restrict__ sm, int E_)
{
    int idx = blockIdx.x * blockDim.x + threadIdx.x;
    if (idx >= E_ * HEADS) return;
    int e = idx >> 3, h = idx & 7;
    int d = dst[e];
    float ex = __expf(logits[idx] - mx[(size_t)d * HEADS + h]);
    logits[idx] = ex;
    atomicAdd(sm + (size_t)d * HEADS + h, ex);
}

__global__ void edge_scatter(const int* __restrict__ src, const int* __restrict__ dst,
                             const float* __restrict__ logits, const float* __restrict__ sm,
                             const float* __restrict__ vt, float* __restrict__ agg, int E_)
{
    int idx = blockIdx.x * blockDim.x + threadIdx.x;
    if (idx >= E_ * HEADS) return;
    int e = idx >> 3, h = idx & 7;
    int s = src[e], d = dst[e];
    float alpha = logits[idx] / (sm[(size_t)d * HEADS + h] + 1e-16f);
    const float* vp = vt + (size_t)s * HID + h * DH;
    float* ap = agg + (size_t)d * HID + h * DH;
    #pragma unroll
    for (int i = 0; i < DH; ++i) atomicAdd(ap + i, alpha * vp[i]);
}

__global__ void gelu_kernel(const float* __restrict__ in, float* __restrict__ out, int n)
{
    int i = blockIdx.x * blockDim.x + threadIdx.x;
    if (i < n) out[i] = gelu_exact(in[i]);
}

// x <- LayerNorm(x + sk*o + (1-sk)*x), sk = sigmoid(skip). One 128-thread block per row.
__global__ void skip_ln(float* __restrict__ x, const float* __restrict__ o,
                        const float* __restrict__ skip_s, const float* __restrict__ g,
                        const float* __restrict__ b)
{
    int row = blockIdx.x, tid = threadIdx.x;
    float xv = x[(size_t)row * HID + tid];
    float ov = o[(size_t)row * HID + tid];
    float sk = sigmoidf_(skip_s[0]);
    float y = xv + sk * ov + (1.0f - sk) * xv;

    __shared__ float red[HID];
    red[tid] = y; __syncthreads();
    for (int s = 64; s > 0; s >>= 1) { if (tid < s) red[tid] += red[tid + s]; __syncthreads(); }
    float mean = red[0] * (1.0f / HID); __syncthreads();
    float dy = y - mean;
    red[tid] = dy * dy; __syncthreads();
    for (int s = 64; s > 0; s >>= 1) { if (tid < s) red[tid] += red[tid + s]; __syncthreads(); }
    float var = red[0] * (1.0f / HID);
    x[(size_t)row * HID + tid] = dy * rsqrtf(var + LN_EPS) * g[tid] + b[tid];
}

extern "C" void kernel_launch(void* const* d_in, const int* in_sizes, int n_in,
                              void* d_out, int out_size, void* d_ws, size_t ws_size,
                              hipStream_t stream)
{
    (void)n_in; (void)out_size; (void)ws_size;
    const float* x_device  = (const float*)d_in[0];
    const float* x_feature = (const float*)d_in[1];
    const int*   e_df_src  = (const int*)d_in[2];
    const int*   e_df_dst  = (const int*)d_in[3];
    const int*   e_fd_src  = (const int*)d_in[4];
    const int*   e_fd_dst  = (const int*)d_in[5];
    const float* W_dev_in  = (const float*)d_in[6];
    const float* b_dev_in  = (const float*)d_in[7];
    const float* W_feat_in = (const float*)d_in[8];
    const float* b_feat_in = (const float*)d_in[9];
    const float* Wk   = (const float*)d_in[10];
    const float* bk   = (const float*)d_in[11];
    const float* Wq   = (const float*)d_in[12];
    const float* bq   = (const float*)d_in[13];
    const float* Wv   = (const float*)d_in[14];
    const float* bv   = (const float*)d_in[15];
    const float* a_rel = (const float*)d_in[16];
    const float* m_rel = (const float*)d_in[17];
    const float* p_rel = (const float*)d_in[18];
    const float* Wa   = (const float*)d_in[19];
    const float* ba   = (const float*)d_in[20];
    const float* skip = (const float*)d_in[21];
    const float* ln_g = (const float*)d_in[22];
    const float* ln_b = (const float*)d_in[23];
    const float* W_out = (const float*)d_in[24];
    const float* b_out = (const float*)d_in[25];

    const int ND_ = in_sizes[0] / HID;
    const int NF_ = in_sizes[1] / HID;
    const int E_  = in_sizes[2];

    // ---- workspace carve (f32 elements) ----
    float* ws = (float*)d_ws;
    size_t off = 0;
    auto alloc = [&](size_t n) { float* r = ws + off; off += n; return r; };
    float* xd   = alloc((size_t)ND_ * HID);
    float* xf   = alloc((size_t)NF_ * HID);
    float* qd   = alloc((size_t)ND_ * HID);
    float* qf   = alloc((size_t)NF_ * HID);
    float* ktd  = alloc((size_t)ND_ * HID);
    float* ktf  = alloc((size_t)NF_ * HID);
    float* vtd  = alloc((size_t)ND_ * HID);
    float* vtf  = alloc((size_t)NF_ * HID);
    float* aggd = alloc((size_t)ND_ * HID);
    float* aggf = alloc((size_t)NF_ * HID);
    float* logits = alloc((size_t)E_ * HEADS);
    float* mx   = alloc((size_t)NF_ * HEADS);
    float* sm   = alloc((size_t)NF_ * HEADS);
    float* Wkf  = alloc(HID * HID);
    float* bkf  = alloc(HID);
    float* Wvf  = alloc(HID * HID);
    float* bvf  = alloc(HID);

    auto gemm = [&](const float* A, const float* B, const float* bias, float* C,
                    int M, int N, int act) {
        int waves  = (M >> 4) * (N >> 4);
        int blocks = (waves * 32 + 255) / 256;
        gemm_bias_act<<<blocks, 256, 0, stream>>>(A, B, bias, C, M, N, act);
    };

    auto edge_pass = [&](const int* esrc, const int* edst, const float* q,
                         const float* ktp, const float* vtp, float* agg, int n_dst) {
        int eh = E_ * HEADS;
        int eb = (eh + 255) / 256;
        int fb = (n_dst * HEADS + 255) / 256;
        hipMemsetAsync(sm, 0, (size_t)n_dst * HEADS * sizeof(float), stream);
        hipMemsetAsync(agg, 0, (size_t)n_dst * HID * sizeof(float), stream);
        fill_kernel<<<fb, 256, 0, stream>>>(mx, -3.4e38f, n_dst * HEADS);
        edge_logits<<<eb, 256, 0, stream>>>(esrc, edst, q, ktp, logits, mx, E_);
        edge_expsum<<<eb, 256, 0, stream>>>(edst, logits, mx, sm, E_);
        edge_scatter<<<eb, 256, 0, stream>>>(esrc, edst, logits, sm, vtp, agg, E_);
    };

    // ---- input projections (ReLU) ----
    gemm(x_device,  W_dev_in,  b_dev_in,  xd, ND_, HID, 1);
    gemm(x_feature, W_feat_in, b_feat_in, xf, NF_, HID, 1);

    const int fblocks = (129 * HID + 255) / 256;

    for (int l = 0; l < LNUM; ++l) {
        for (int t = 0; t < 2; ++t) {
            const int lt = l * 2 + t;
            // fold a_rel*p/sqrtD into K weights, m_rel into V weights
            fuse_rel<<<fblocks, 256, 0, stream>>>(Wk + (size_t)lt * HID * HID, bk + lt * HID,
                                                  a_rel + (size_t)lt * HEADS * DH * DH,
                                                  p_rel + lt * HEADS, Wkf, bkf);
            fuse_rel<<<fblocks, 256, 0, stream>>>(Wv + (size_t)lt * HID * HID, bv + lt * HID,
                                                  m_rel + (size_t)lt * HEADS * DH * DH,
                                                  (const float*)nullptr, Wvf, bvf);
            const float* xt = t ? xf : xd;
            const int    Nt = t ? NF_ : ND_;
            gemm(xt, Wq + (size_t)lt * HID * HID, bq + lt * HID, t ? qf : qd,  Nt, HID, 0);
            gemm(xt, Wkf, bkf,                                   t ? ktf : ktd, Nt, HID, 0);
            gemm(xt, Wvf, bvf,                                   t ? vtf : vtd, Nt, HID, 0);
        }
        // edge type 0: device -> feature ; edge type 1: feature -> device
        edge_pass(e_df_src, e_df_dst, qf, ktd, vtd, aggf, NF_);
        edge_pass(e_fd_src, e_fd_dst, qd, ktf, vtf, aggd, ND_);

        // gelu -> Wa -> skip + residual + LN (kt buffers reused as gelu scratch,
        // q buffers reused as Wa output)
        int nd_el = ND_ * HID, nf_el = NF_ * HID;
        gelu_kernel<<<(nd_el + 255) / 256, 256, 0, stream>>>(aggd, ktd, nd_el);
        gelu_kernel<<<(nf_el + 255) / 256, 256, 0, stream>>>(aggf, ktf, nf_el);
        gemm(ktd, Wa + (size_t)(l * 2 + 0) * HID * HID, ba + (l * 2 + 0) * HID, qd, ND_, HID, 0);
        gemm(ktf, Wa + (size_t)(l * 2 + 1) * HID * HID, ba + (l * 2 + 1) * HID, qf, NF_, HID, 0);
        skip_ln<<<ND_, HID, 0, stream>>>(xd, qd, skip + l * 2 + 0,
                                         ln_g + (size_t)(l * 2 + 0) * HID,
                                         ln_b + (size_t)(l * 2 + 0) * HID);
        skip_ln<<<NF_, HID, 0, stream>>>(xf, qf, skip + l * 2 + 1,
                                         ln_g + (size_t)(l * 2 + 1) * HID,
                                         ln_b + (size_t)(l * 2 + 1) * HID);
    }

    // ---- output head: [ND,128] @ [128,32] ----
    gemm(xd, W_out, b_out, (float*)d_out, ND_, OUTD, 0);
}
